// PhimoeSparseMoeBlock_6975026888748
// MI455X (gfx1250) — matrix operand order repaired and checked by reading, not compiled, not run on hardware
//
#include <hip/hip_runtime.h>
#include <hip/hip_bf16.h>

namespace {

constexpr int E = 8, H = 1024, F = 2048, T = 1024;
constexpr float JITTER = 0.01f;

typedef __attribute__((ext_vector_type(16))) __bf16 v16bf;
typedef __attribute__((ext_vector_type(8)))  float  v8f;
typedef __attribute__((ext_vector_type(4)))  int    v4i;

union FragB16 {
  v16bf v;
  unsigned short u[16];
  uint4 q[2];            // two 16-byte chunks
};

// ---- fp32 -> bf16 conversion helpers (prefer packed HW cvt) ----------------
__device__ __forceinline__ unsigned int pack2_bf16(float a, float b) {
#if __has_builtin(__builtin_amdgcn_cvt_pk_bf16_f32)
  typedef __attribute__((ext_vector_type(2))) __bf16 v2bf;
  v2bf p = __builtin_amdgcn_cvt_pk_bf16_f32(a, b);
  return __builtin_bit_cast(unsigned int, p);
#else
  // gfx1250 packed convert (2 fp32 -> 2 bf16 in one VALU op)
  unsigned int r;
  asm("v_cvt_pk_bf16_f32 %0, %1, %2" : "=v"(r) : "v"(a), "v"(b));
  return r;
#endif
}

__device__ __forceinline__ uint2 pack4_bf16(float4 v) {
  return make_uint2(pack2_bf16(v.x, v.y), pack2_bf16(v.z, v.w));
}

__device__ __forceinline__ unsigned short bf16_bits(float f) {
  union { __bf16 h; unsigned short u; } c;
  c.h = (__bf16)f;
  return c.u;
}

__device__ __forceinline__ float fast_rcp(float x) {
#if __has_builtin(__builtin_amdgcn_rcpf)
  return __builtin_amdgcn_rcpf(x);   // v_rcp_f32
#else
  return 1.f / x;
#endif
}

// ---- async global->LDS copy (gfx1250 async path), guarded ------------------
#if __has_builtin(__builtin_amdgcn_global_load_async_to_lds_b128)
#define ASYNC_LDS_COPY 1
#else
#define ASYNC_LDS_COPY 0
#endif

__device__ __forceinline__ void lds_copy_b128(const void* gsrc, void* ldst) {
#if ASYNC_LDS_COPY
  __builtin_amdgcn_global_load_async_to_lds_b128((v4i*)gsrc, (v4i*)ldst, 0, 0);
#else
  *(uint4*)ldst = *(const uint4*)gsrc;
#endif
}

__device__ __forceinline__ void wait_asynccnt0() {
#if ASYNC_LDS_COPY
#if __has_builtin(__builtin_amdgcn_s_wait_asynccnt)
  __builtin_amdgcn_s_wait_asynccnt(0);
#else
  asm volatile("s_wait_asynccnt 0x0" ::: "memory");
#endif
#endif
}

// ---------------------------------------------------------------------------
// Kernel 1: router logits + sparsemixer(eval) weights [T,E]; also converts
//           x -> bf16 once (reused 512x by kernel 2 -> pure-copy staging)
// ---------------------------------------------------------------------------
__global__ void router_kernel(const float* __restrict__ x,
                              const float* __restrict__ gate_w,
                              float* __restrict__ out_logits,
                              float* __restrict__ weight,
                              unsigned short* __restrict__ xb16) {
  __shared__ float g[E * H];              // 32 KB gate weights
  const int tid = threadIdx.x;
  for (int i = tid; i < E * H; i += 256) g[i] = gate_w[i];

  // convert this block's 256 token rows of x to bf16 (coalesced, no sync dep)
  {
    const size_t base = (size_t)blockIdx.x * 256 * H;
    for (int i = tid * 2; i < 256 * H; i += 512) {
      float2 v = *(const float2*)(x + base + i);
      *(unsigned int*)(xb16 + base + i) = pack2_bf16(v.x, v.y);
    }
  }
  __syncthreads();

  const int t = blockIdx.x * 256 + tid;
  const float* xr = x + (size_t)t * H;

  float s[E];
#pragma unroll
  for (int e = 0; e < E; ++e) s[e] = 0.f;
  for (int h = 0; h < H; h += 4) {
    float4 xv = *(const float4*)(xr + h);
#pragma unroll
    for (int e = 0; e < E; ++e) {
      const float* gr = g + e * H + h;
      s[e] += xv.x * gr[0] + xv.y * gr[1] + xv.z * gr[2] + xv.w * gr[3];
    }
  }

  // ---- top-1 ----
  float max1 = s[0]; int sel1 = 0;
#pragma unroll
  for (int e = 1; e < E; ++e) if (s[e] > max1) { max1 = s[e]; sel1 = e; }
  float den1 = 0.f;
#pragma unroll
  for (int e = 0; e < E; ++e) {
    float factor = fmaxf(fabsf(s[e]), max1);
    bool masked = (max1 - s[e]) > 2.f * JITTER * factor;  // factor >= 0
    if (!masked) den1 += __expf(s[e] - max1);
  }
  float mult1 = fast_rcp(den1);            // exp(0) / den1

  // ---- top-2 (top-1 removed) ----
  float max2 = -__builtin_huge_valf(); int sel2 = 0;
#pragma unroll
  for (int e = 0; e < E; ++e)
    if (e != sel1 && s[e] > max2) { max2 = s[e]; sel2 = e; }
  float den2 = 0.f;
#pragma unroll
  for (int e = 0; e < E; ++e) {
    if (e == sel1) continue;
    float factor = fmaxf(fabsf(s[e]), max2);
    bool masked = (max2 - s[e]) > 2.f * JITTER * factor;
    if (!masked) den2 += __expf(s[e] - max2);
  }
  float mult2 = fast_rcp(den2);

#pragma unroll
  for (int e = 0; e < E; ++e) {
    out_logits[(size_t)t * E + e] = s[e];
    float w = (e == sel1 ? mult1 : 0.f) + (e == sel2 ? mult2 : 0.f);
    weight[(size_t)t * E + e] = w;
  }
}

// ---------------------------------------------------------------------------
// Kernel 2: per-expert up-projection, fused SwiGLU -> inter[E,T,F] (bf16)
//   block tile: 64 tokens x 32 F-cols, 8 waves = 4(M) x 2(N) WMMA 16x16
//   A tile: async bf16 copy; B tiles (w1/w3): fp32 load + packed cvt in LDS
// ---------------------------------------------------------------------------
__global__ void moe_up_kernel(const unsigned short* __restrict__ xb16,
                              const float* __restrict__ w1,
                              const float* __restrict__ w3,
                              unsigned short* __restrict__ inter) {
  __shared__ __align__(16) unsigned short ldsA [64][40];  // x tile   (bf16)
  __shared__ __align__(16) unsigned short ldsB1[32][40];  // w1 tile  (bf16)
  __shared__ __align__(16) unsigned short ldsB3[32][40];  // w3 tile  (bf16)

  const int e  = blockIdx.z;
  const int fB = blockIdx.x * 32;
  const int tB = blockIdx.y * 64;
  const int tid  = threadIdx.x;
  const int lane = tid & 31;
  const int wave = tid >> 5;
  const int wm = wave & 3;           // M sub-tile (16 tokens each)
  const int wn = wave >> 2;          // N sub-tile (16 F-cols each)
  const int col  = lane & 15;
  const int half = lane >> 4;
  const int ncol = wn * 16 + col;    // local N column 0..31
  const int kq   = half * 8;         // B frag: lanes0-15 K0-7,16-23, lanes16-31 K8-15,24-31

  v8f acc1 = {}; v8f acc3 = {};

  const int arow = tid >> 2, ac = tid & 3;   // A-tile copy: 256 x b128
  const int brow = tid >> 3, bc4 = tid & 7;  // B-tile staging rows / K slots

  for (int k0 = 0; k0 < H; k0 += 32) {
    // ---- stage x[tB..+63][k0..+31] (pure bf16 copy, async when available) --
    lds_copy_b128(xb16 + (size_t)(tB + arow) * H + k0 + ac * 8,
                  &ldsA[arow][ac * 8]);
    // ---- stage w1/w3 tiles [32 fcols x 32 K] (1 float4 each / thread) ----
    {
      const float* p1 = w1 + ((size_t)e * F + fB + brow) * H + k0 + bc4 * 4;
      const float* p3 = w3 + ((size_t)e * F + fB + brow) * H + k0 + bc4 * 4;
      __builtin_prefetch(p1 + 32, 0, 1);     // global_prefetch next K-step
      __builtin_prefetch(p3 + 32, 0, 1);
      *(uint2*)&ldsB1[brow][bc4 * 4] = pack4_bf16(*(const float4*)p1);
      *(uint2*)&ldsB3[brow][bc4 * 4] = pack4_bf16(*(const float4*)p3);
    }
    wait_asynccnt0();
    __syncthreads();

    // A fragment: per lane K chunks {koff..+7} and {koff+16..+23}
    FragB16 a, b1, b3;
    {
      int row  = wm * 16 + col;
      int koff = half * 8;
      a.q[0] = *(const uint4*)&ldsA[row][koff];
      a.q[1] = *(const uint4*)&ldsA[row][koff + 16];
      b1.q[0] = *(const uint4*)&ldsB1[ncol][kq];
      b1.q[1] = *(const uint4*)&ldsB1[ncol][kq + 16];
      b3.q[0] = *(const uint4*)&ldsB3[ncol][kq];
      b3.q[1] = *(const uint4*)&ldsB3[ncol][kq + 16];
    }
    acc1 = __builtin_amdgcn_wmma_f32_16x16x32_bf16(false, a.v, false, b1.v,
                                                   (short)0, acc1, false, false);
    acc3 = __builtin_amdgcn_wmma_f32_16x16x32_bf16(false, a.v, false, b3.v,
                                                   (short)0, acc3, false, false);
    __syncthreads();
  }

  // inter = silu(h1) * h3, stored bf16. C/D layout: row m = r + half*8, col n
#pragma unroll
  for (int r = 0; r < 8; ++r) {
    int t = tB + wm * 16 + r + half * 8;
    float a1 = acc1[r];
    float sv = a1 * fast_rcp(1.f + __expf(-a1));
    float iv = sv * acc3[r];
    inter[((size_t)e * T + t) * F + fB + ncol] = bf16_bits(iv);
  }
}

// ---------------------------------------------------------------------------
// Kernel 3: down-projection + weighted combine over all experts
//   block tile: 64 tokens x 32 H-cols; loops experts in-block (no atomics)
// ---------------------------------------------------------------------------
__global__ void moe_down_kernel(const unsigned short* __restrict__ inter,
                                const float* __restrict__ w2,
                                const float* __restrict__ weight,
                                float* __restrict__ out) {
  __shared__ __align__(16) unsigned short ldsA[64][40];  // inter tile (bf16)
  __shared__ __align__(16) unsigned short ldsB[32][40];  // w2 tile    (bf16)
  __shared__ float ldsW[64][8];                          // routing weights

  const int hB = blockIdx.x * 32;
  const int tB = blockIdx.y * 64;
  const int tid  = threadIdx.x;
  const int lane = tid & 31;
  const int wave = tid >> 5;
  const int wm = wave & 3;
  const int wn = wave >> 2;
  const int col  = lane & 15;
  const int half = lane >> 4;
  const int ncol = wn * 16 + col;
  const int kq   = half * 8;

  for (int i = tid; i < 64 * 8; i += 256)
    (&ldsW[0][0])[i] = weight[(size_t)(tB + (i >> 3)) * E + (i & 7)];

  float facc[8];
#pragma unroll
  for (int r = 0; r < 8; ++r) facc[r] = 0.f;

  const int arow = tid >> 2, ac = tid & 3;   // A-tile copy: 256 x b128
  const int brow = tid >> 3, bc4 = tid & 7;  // B-tile staging: 256 x float4

  for (int e = 0; e < E; ++e) {
    v8f acc = {};
    const unsigned short* ip = inter + (size_t)e * T * F;

    for (int k0 = 0; k0 < F; k0 += 32) {
      __syncthreads();
      // ---- stage inter[tB..+63][k0..+31] (pure bf16 copy) ----
      lds_copy_b128(ip + (size_t)(tB + arow) * F + k0 + ac * 8,
                    &ldsA[arow][ac * 8]);
      // ---- stage w2 tile [32 hcols x 32 K], convert fp32->bf16 ----
      {
        const float* p = w2 + ((size_t)e * H + hB + brow) * F + k0 + bc4 * 4;
        __builtin_prefetch(p + 32, 0, 1);
        *(uint2*)&ldsB[brow][bc4 * 4] = pack4_bf16(*(const float4*)p);
      }
      wait_asynccnt0();
      __syncthreads();

      FragB16 a, b;
      {
        int row  = wm * 16 + col;
        int koff = half * 8;
        a.q[0] = *(const uint4*)&ldsA[row][koff];
        a.q[1] = *(const uint4*)&ldsA[row][koff + 16];
        b.q[0] = *(const uint4*)&ldsB[ncol][kq];
        b.q[1] = *(const uint4*)&ldsB[ncol][kq + 16];
      }
      acc = __builtin_amdgcn_wmma_f32_16x16x32_bf16(false, a.v, false, b.v,
                                                    (short)0, acc, false, false);
    }
    // scale by routing weight of this token row and accumulate
#pragma unroll
    for (int r = 0; r < 8; ++r) {
      int tl = wm * 16 + r + half * 8;
      facc[r] += ldsW[tl][e] * acc[r];
    }
  }

#pragma unroll
  for (int r = 0; r < 8; ++r) {
    int t = tB + wm * 16 + r + half * 8;
    out[(size_t)t * H + hB + ncol] = facc[r];
  }
}

}  // namespace

extern "C" void kernel_launch(void* const* d_in, const int* in_sizes, int n_in,
                              void* d_out, int out_size, void* d_ws, size_t ws_size,
                              hipStream_t stream) {
  (void)in_sizes; (void)n_in; (void)out_size; (void)ws_size;
  const float* x      = (const float*)d_in[0];   // [1,1024,1024]
  const float* gate_w = (const float*)d_in[1];   // [8,1024]
  const float* w1     = (const float*)d_in[2];   // [8,2048,1024]
  const float* w2     = (const float*)d_in[3];   // [8,1024,2048]
  const float* w3     = (const float*)d_in[4];   // [8,2048,1024]

  float* outF   = (float*)d_out;                 // final [T,H]
  float* logits = outF + (size_t)T * H;          // router_logits [T,E]

  float* weight = (float*)d_ws;                                      // [T,E] fp32 (32 KB)
  unsigned short* xb16  = (unsigned short*)((char*)d_ws + 32768);    // [T,H] bf16 (2 MB)
  unsigned short* inter = (unsigned short*)((char*)d_ws + 32768 + (size_t)T * H * 2);
                                                                     // [E,T,F] bf16 (~33.6 MB)

  router_kernel<<<dim3(T / 256), dim3(256), 0, stream>>>(x, gate_w, logits, weight, xb16);
  moe_up_kernel<<<dim3(F / 32, T / 64, E), dim3(256), 0, stream>>>(xb16, w1, w3, inter);
  moe_down_kernel<<<dim3(H / 32, T / 64), dim3(256), 0, stream>>>(inter, w2, weight, outF);
}
